// Detect_39135742001455
// MI455X (gfx1250) — matrix operand order repaired and checked by reading, not compiled, not run on hardware
//
#include <hip/hip_runtime.h>
#include <hip/hip_bf16.h>

typedef __bf16 bf16_t;
typedef __attribute__((ext_vector_type(16))) __bf16 v16bf;
typedef __attribute__((ext_vector_type(8)))  __bf16 v8bf;
typedef __attribute__((ext_vector_type(8)))  float  v8f;

#define NA_ 3
#define HD_ 512
#define NO_ 5
#define NT_ 16
#define TD_ 512

// ---------------- Stage 1: qw[(t*5+j), d] = (target@q_w^T + q_b)[t,d] * out_w[j,d] (bf16)
__global__ void build_qw_kernel(const float* __restrict__ target,
                                const float* __restrict__ q_w,
                                const float* __restrict__ q_b,
                                const float* __restrict__ out_w,
                                bf16_t* __restrict__ qw) {
    const int t = blockIdx.x;
    for (int d = threadIdx.x; d < HD_; d += blockDim.x) {
        float acc = q_b[d];
        const float* tr = target + t * TD_;
        const float* qr = q_w + (size_t)d * TD_;
        for (int k = 0; k < TD_; ++k) acc += tr[k] * qr[k];
        for (int j = 0; j < NO_; ++j)
            qw[(size_t)(t * NO_ + j) * HD_ + d] = (bf16_t)(acc * out_w[j * HD_ + d]);
    }
}

// ---------------- f32 -> bf16 weight pre-pack
__global__ void pack_bf16_kernel(const float* __restrict__ src,
                                 bf16_t* __restrict__ dst, int n) {
    int i = blockIdx.x * blockDim.x + threadIdx.x;
    if (i < n) dst[i] = (bf16_t)src[i];
}

// load 8 consecutive weights (f32 with cvt, or bf16 direct) into dst[base..base+7]
template <typename WT>
static __device__ inline void loadA8(const WT* __restrict__ p, v16bf& dst, int base) {
    if constexpr (__is_same(WT, float)) {
        float4 f0 = ((const float4*)p)[0];
        float4 f1 = ((const float4*)p)[1];
        dst[base + 0] = (bf16_t)f0.x; dst[base + 1] = (bf16_t)f0.y;
        dst[base + 2] = (bf16_t)f0.z; dst[base + 3] = (bf16_t)f0.w;
        dst[base + 4] = (bf16_t)f1.x; dst[base + 5] = (bf16_t)f1.y;
        dst[base + 6] = (bf16_t)f1.z; dst[base + 7] = (bf16_t)f1.w;
    } else {
        v8bf t = *(const v8bf*)p;           // one 16B load
        #pragma unroll
        for (int i = 0; i < 8; ++i) dst[base + i] = t[i];
    }
}

struct Params {
    const float *x0, *x1, *x2;
    const void  *kw0, *vw0, *kw1, *vw1, *kw2, *vw2;
    const float *kb0, *vb0, *kb1, *vb1, *kb2, *vb2;
    const bf16_t* qw;
    const float*  outb;
    float*        out;
};

// ---------------- Fused stage 2, all 3 levels in one grid.
// Block = (level, batch b, 128-pixel tile). x tile staged in LDS bf16 [128][C+8].
// 12 waves per block: wave w -> anchor a = w>>2, m-quarter mQ = w&3.
// Each wave: FULL d loop (0..511), 32 pixels (two 16m sub-tiles), 80 (t,j) cols.
template <typename WT>
__global__ __launch_bounds__(384)
void fused_kernel(Params p) {
    extern __shared__ char smem_raw[];
    bf16_t* xs = (bf16_t*)smem_raw;            // [128][C+8]

    // --- block -> level / batch / pixel tile (all wave-uniform scalar work)
    int blk = blockIdx.x;
    const float* x; const WT* kw; const WT* vw; const float* kb; const float* vb;
    float* out; int C, P, b, mBase;
    if (blk < 64) {              // level 0: 2 b x 32 tiles
        C = 256;  P = 4096; b = blk >> 5; mBase = (blk & 31) << 7;
        x = p.x0; kw = (const WT*)p.kw0; vw = (const WT*)p.vw0; kb = p.kb0; vb = p.vb0;
        out = p.out;
    } else if (blk < 80) {       // level 1: 2 b x 8 tiles
        blk -= 64;
        C = 512;  P = 1024; b = blk >> 3; mBase = (blk & 7) << 7;
        x = p.x1; kw = (const WT*)p.kw1; vw = (const WT*)p.vw1; kb = p.kb1; vb = p.vb1;
        out = p.out + 1966080;
    } else {                     // level 2: 2 b x 2 tiles
        blk -= 80;
        C = 1024; P = 256;  b = blk >> 1; mBase = (blk & 1) << 7;
        x = p.x2; kw = (const WT*)p.kw2; vw = (const WT*)p.vw2; kb = p.kb2; vb = p.vb2;
        out = p.out + 2457600;
    }
    const int ldx = C + 8;
    const float* xb = x + (size_t)b * C * P;

    // --- cooperative LDS fill: coalesced global reads (128 consecutive pixels per c),
    // bf16 convert, transposed store xs[mi][c].
    const int tid = threadIdx.x;
    for (int idx = tid; idx < (C << 7); idx += 384) {
        const int c = idx >> 7, mi = idx & 127;
        xs[mi * ldx + c] = (bf16_t)xb[(size_t)c * P + mBase + mi];
    }
    __syncthreads();

    const int lane = tid & 31;
    const int w    = tid >> 5;          // 0..11
    const int a    = w >> 2;            // anchor
    const int mLoc = (w & 3) << 5;      // wave's 32-pixel sub-block within the 128 tile
    const int row  = lane & 15;
    const int hi   = lane >> 4;

    v8f racc[2][5];
    #pragma unroll
    for (int s = 0; s < 2; ++s)
        #pragma unroll
        for (int i = 0; i < 5; ++i)
            #pragma unroll
            for (int g = 0; g < 8; ++g) racc[s][i][g] = 0.0f;

    #pragma unroll 1
    for (int d0 = 0; d0 < HD_; d0 += 32) {
        v16bf a2[2];   // kv tiles (16m x 32d bf16) = GEMM2 A operands
        #pragma unroll
        for (int half = 0; half < 2; ++half) {
            const int dRowBase = a * HD_ + d0 + half * 16;
            v8f kacc[2], vacc[2];
            #pragma unroll
            for (int s = 0; s < 2; ++s)
                #pragma unroll
                for (int g = 0; g < 8; ++g) { kacc[s][g] = 0.0f; vacc[s][g] = 0.0f; }

            for (int c0 = 0; c0 < C; c0 += 32) {
                // A operands (16d x 32c) from weights (shared stream across mQ waves -> WGP$ hits)
                const WT* kwp = kw + (size_t)(dRowBase + row) * C + c0 + hi * 8;
                const WT* vwp = vw + (size_t)(dRowBase + row) * C + c0 + hi * 8;
                v16bf ak, av;
                loadA8(kwp,      ak, 0); loadA8(kwp + 16, ak, 8);
                loadA8(vwp,      av, 0); loadA8(vwp + 16, av, 8);
                // B operands (32c x 16m) from LDS: two 16B ds loads per tile
                const bf16_t* xp0 = xs + (mLoc + row) * ldx + c0 + hi * 8;
                const bf16_t* xp1 = xs + (mLoc + 16 + row) * ldx + c0 + hi * 8;
                v16bf bx0, bx1;
                {
                    v8bf h0 = *(const v8bf*)xp0, h1 = *(const v8bf*)(xp0 + 16);
                    v8bf g0 = *(const v8bf*)xp1, g1 = *(const v8bf*)(xp1 + 16);
                    #pragma unroll
                    for (int i = 0; i < 8; ++i) {
                        bx0[i] = h0[i]; bx0[i + 8] = h1[i];
                        bx1[i] = g0[i]; bx1[i + 8] = g1[i];
                    }
                }
                kacc[0] = __builtin_amdgcn_wmma_f32_16x16x32_bf16(false, ak, false, bx0,
                                                                  (short)0, kacc[0], false, false);
                kacc[1] = __builtin_amdgcn_wmma_f32_16x16x32_bf16(false, ak, false, bx1,
                                                                  (short)0, kacc[1], false, false);
                vacc[0] = __builtin_amdgcn_wmma_f32_16x16x32_bf16(false, av, false, bx0,
                                                                  (short)0, vacc[0], false, false);
                vacc[1] = __builtin_amdgcn_wmma_f32_16x16x32_bf16(false, av, false, bx1,
                                                                  (short)0, vacc[1], false, false);
            }
            // biases + kv; D1 VGPR g holds d = dRowBase + hi*8 + g.
            const float* kbp = kb + dRowBase + hi * 8;
            const float* vbp = vb + dRowBase + hi * 8;
            float4 kb0 = ((const float4*)kbp)[0], kb1 = ((const float4*)kbp)[1];
            float4 vb0 = ((const float4*)vbp)[0], vb1 = ((const float4*)vbp)[1];
            float kbl[8] = {kb0.x, kb0.y, kb0.z, kb0.w, kb1.x, kb1.y, kb1.z, kb1.w};
            float vbl[8] = {vb0.x, vb0.y, vb0.z, vb0.w, vb1.x, vb1.y, vb1.z, vb1.w};
            #pragma unroll
            for (int s = 0; s < 2; ++s)
                #pragma unroll
                for (int g = 0; g < 8; ++g) {
                    float kv = (kacc[s][g] + kbl[g]) * (vacc[s][g] + vbl[g]);
                    // D1 register layout == GEMM2 A-matrix layout: pack directly.
                    a2[s][half * 8 + g] = (bf16_t)kv;
                }
        }
        // GEMM2: r[32m x 80tj] += a2 x qw^T ; B tile shared by both m-sub-tiles.
        #pragma unroll
        for (int tjt = 0; tjt < 5; ++tjt) {
            const bf16_t* qp = p.qw + (size_t)(tjt * 16 + row) * HD_ + d0 + hi * 8;
            v8bf q0 = *(const v8bf*)qp, q1 = *(const v8bf*)(qp + 16);
            v16bf b2 = __builtin_shufflevector(q0, q1, 0, 1, 2, 3, 4, 5, 6, 7, 8, 9, 10, 11, 12, 13, 14, 15);
            racc[0][tjt] = __builtin_amdgcn_wmma_f32_16x16x32_bf16(false, a2[0], false, b2,
                                                                   (short)0, racc[0][tjt], false, false);
            racc[1][tjt] = __builtin_amdgcn_wmma_f32_16x16x32_bf16(false, a2[1], false, b2,
                                                                   (short)0, racc[1][tjt], false, false);
        }
    }

    // Epilogue: D2 VGPR g -> m = mBase + mLoc + s*16 + g + hi*8, lane -> tj = tjt*16 + row.
    #pragma unroll
    for (int tjt = 0; tjt < 5; ++tjt) {
        const int tj = tjt * 16 + row;
        const int t  = tj / NO_;
        const int j  = tj % NO_;
        const float bias = p.outb[j];
        #pragma unroll
        for (int s = 0; s < 2; ++s)
            #pragma unroll
            for (int g = 0; g < 8; ++g) {
                const int m = mBase + mLoc + s * 16 + g + hi * 8;
                const size_t off = ((((size_t)b * NT_ + t) * NA_ + a) * P + m) * NO_ + j;
                out[off] = racc[s][tjt][g] + bias;
            }
    }
}

extern "C" void kernel_launch(void* const* d_in, const int* in_sizes, int n_in,
                              void* d_out, int out_size, void* d_ws, size_t ws_size,
                              hipStream_t stream) {
    (void)in_sizes; (void)n_in; (void)out_size;
    const float* target = (const float*)d_in[3];
    const float* q_w  = (const float*)d_in[16];
    const float* q_b  = (const float*)d_in[17];
    const float* outw = (const float*)d_in[18];

    bf16_t* qw = (bf16_t*)d_ws;                      // 80x512 bf16 = 80 KB
    build_qw_kernel<<<NT_, 256, 0, stream>>>(target, q_w, q_b, outw, qw);

    Params p;
    p.x0 = (const float*)d_in[0]; p.x1 = (const float*)d_in[1]; p.x2 = (const float*)d_in[2];
    p.kb0 = (const float*)d_in[5];  p.vb0 = (const float*)d_in[7];
    p.kb1 = (const float*)d_in[9];  p.vb1 = (const float*)d_in[11];
    p.kb2 = (const float*)d_in[13]; p.vb2 = (const float*)d_in[15];
    p.qw = qw;
    p.outb = (const float*)d_in[19];
    p.out  = (float*)d_out;

    const int nW0 = NA_ * HD_ * 256, nW1 = NA_ * HD_ * 512, nW2 = NA_ * HD_ * 1024;
    const size_t qwElems = (size_t)NT_ * NO_ * HD_;
    const size_t needBytes = (qwElems + 2ull * (nW0 + nW1 + nW2)) * sizeof(bf16_t);
    const bool wbf = ws_size >= needBytes;

    const size_t smem = 128 * (1024 + 8) * sizeof(bf16_t);   // 264 KB (level-2 tile)
    const int grid = 64 + 16 + 4;

    if (wbf) {
        bf16_t* wsp  = qw + qwElems;
        bf16_t* kwb0 = wsp;          bf16_t* vwb0 = kwb0 + nW0;
        bf16_t* kwb1 = vwb0 + nW0;   bf16_t* vwb1 = kwb1 + nW1;
        bf16_t* kwb2 = vwb1 + nW1;   bf16_t* vwb2 = kwb2 + nW2;
        pack_bf16_kernel<<<(nW0 + 255) / 256, 256, 0, stream>>>((const float*)d_in[4],  kwb0, nW0);
        pack_bf16_kernel<<<(nW0 + 255) / 256, 256, 0, stream>>>((const float*)d_in[6],  vwb0, nW0);
        pack_bf16_kernel<<<(nW1 + 255) / 256, 256, 0, stream>>>((const float*)d_in[8],  kwb1, nW1);
        pack_bf16_kernel<<<(nW1 + 255) / 256, 256, 0, stream>>>((const float*)d_in[10], vwb1, nW1);
        pack_bf16_kernel<<<(nW2 + 255) / 256, 256, 0, stream>>>((const float*)d_in[12], kwb2, nW2);
        pack_bf16_kernel<<<(nW2 + 255) / 256, 256, 0, stream>>>((const float*)d_in[14], vwb2, nW2);
        p.kw0 = kwb0; p.vw0 = vwb0; p.kw1 = kwb1; p.vw1 = vwb1; p.kw2 = kwb2; p.vw2 = vwb2;
        fused_kernel<bf16_t><<<grid, 384, smem, stream>>>(p);
    } else {
        p.kw0 = d_in[4];  p.vw0 = d_in[6];
        p.kw1 = d_in[8];  p.vw1 = d_in[10];
        p.kw2 = d_in[12]; p.vw2 = d_in[14];
        fused_kernel<float><<<grid, 384, smem, stream>>>(p);
    }
}
